// DepartmentClassifierRNN_13511967113667
// MI455X (gfx1250) — matrix operand order repaired and checked by reading, not compiled, not run on hardware
//
#include <hip/hip_runtime.h>
#include <hip/hip_bf16.h>
#include <math.h>

// Problem constants (match reference)
#define Vv 32000
#define Ee 512
#define Hh 512
#define Oo 32
#define Ll 2
#define Bb 64
#define Tt 512

#define NWG 16          // workgroups; each owns 32 output columns of H=512
#define NTHREADS 256    // 8 waves of 32
#define WSZ (Hh * Hh)   // 262144 elements per 512x512 matrix

typedef __bf16  v16bf __attribute__((ext_vector_type(16)));
typedef __bf16  v8bf  __attribute__((ext_vector_type(8)));
typedef float   v8f   __attribute__((ext_vector_type(8)));
typedef float   f32x8 __attribute__((ext_vector_type(8)));

// ---------------------------------------------------------------------------
// WMMA helper: D = A(16x32 bf16) * B(32x16 bf16) + C(16x16 f32)
// ---------------------------------------------------------------------------
__device__ __forceinline__ v8f wmma_bf16(v16bf a, v16bf b, v8f c) {
  return __builtin_amdgcn_wmma_f32_16x16x32_bf16(
      /*neg_a=*/false, a, /*neg_b=*/false, b,
      /*c_mod=*/(short)0, c, /*reuse_a=*/false, /*reuse_b=*/false);
}

// A fragment (16-bit A-matrix 16x32 layout, ISA 7.12.2):
// lane m=lane&15, khalf=lane>>4; elems 0..7 -> K = khalf*8 + i,
// elems 8..15 -> K = 16 + khalf*8 + (i-8). Two contiguous 16B chunks.
__device__ __forceinline__ v16bf load_a_bf16(const __bf16* row, int k0, int khalf) {
  v8bf a0 = *(const v8bf*)(row + k0 + khalf * 8);
  v8bf a1 = *(const v8bf*)(row + k0 + 16 + khalf * 8);
  v16bf a;
#pragma unroll
  for (int i = 0; i < 8; ++i) { a[i] = a0[i]; a[8 + i] = a1[i]; }
  return a;
}

// Same A layout but gathering from an f32 row (embedding table) with cvt.
__device__ __forceinline__ v16bf load_a_f32(const float* row, int k0, int khalf) {
  f32x8 a0 = *(const f32x8*)(row + k0 + khalf * 8);
  f32x8 a1 = *(const f32x8*)(row + k0 + 16 + khalf * 8);
  v16bf a;
#pragma unroll
  for (int i = 0; i < 8; ++i) { a[i] = (__bf16)a0[i]; a[8 + i] = (__bf16)a1[i]; }
  return a;
}

// B fragment (32x16 bf16): lane n=lane&15, khalf=lane>>4;
// elems 0..7 -> K = khalf*8 + i, elems 8..15 -> K = 16 + khalf*8 + (i-8) (same as A).
__device__ __forceinline__ v16bf load_b_lds(const __bf16* wrow, int k0, int khalf) {
  v8bf b0 = *(const v8bf*)(wrow + k0 + khalf * 8);
  v8bf b1 = *(const v8bf*)(wrow + k0 + 16 + khalf * 8);
  v16bf b;
#pragma unroll
  for (int i = 0; i < 8; ++i) { b[i] = b0[i]; b[8 + i] = b1[i]; }
  return b;
}

// ---------------------------------------------------------------------------
// Grid-wide barrier (monotonic generation counter). 16 WGs are co-resident.
// ---------------------------------------------------------------------------
__device__ __forceinline__ void grid_barrier(int* cnt, int target) {
  __syncthreads();
  if (threadIdx.x == 0) {
    __hip_atomic_fetch_add(cnt, 1, __ATOMIC_RELEASE, __HIP_MEMORY_SCOPE_AGENT);
    while (__hip_atomic_load(cnt, __ATOMIC_ACQUIRE, __HIP_MEMORY_SCOPE_AGENT) < target) {
      __builtin_amdgcn_s_sleep(2);
    }
  }
  __syncthreads();
}

// ---------------------------------------------------------------------------
// Prologue kernels
// ---------------------------------------------------------------------------
__global__ void cvt_f32_to_bf16(const float* __restrict__ src,
                                __bf16* __restrict__ dst, int n) {
  int i = blockIdx.x * blockDim.x + threadIdx.x;
  int s = gridDim.x * blockDim.x;
  for (; i < n; i += s) dst[i] = (__bf16)src[i];
}

__global__ void zero_bytes16(char* __restrict__ p, int nchunks16) {
  int i = blockIdx.x * blockDim.x + threadIdx.x;
  int s = gridDim.x * blockDim.x;
  int4 z = make_int4(0, 0, 0, 0);
  for (; i < nchunks16; i += s) ((int4*)p)[i] = z;
}

// ---------------------------------------------------------------------------
// Persistent RNN kernel.
// Grid: 16 WGs x 256 threads. WG wg owns output columns [wg*32, wg*32+32).
// LDS: 5 weight slices [32 rows x 512 k] bf16 = 160 KB (resident all steps).
//   mtx 0: Whx layer0   1: Whh layer0   2: Whx layer1   3: Whh layer1   4: Wyh[1]
// Hidden state double-buffered in global ws; one grid barrier per timestep.
// ---------------------------------------------------------------------------
__global__ __launch_bounds__(NTHREADS, 1)
void rnn_persistent(const int* __restrict__ x, const int* __restrict__ slen,
                    const float* __restrict__ emb,
                    const float* __restrict__ b_h, const float* __restrict__ b_y,
                    const float* __restrict__ Wf, const float* __restrict__ bfv,
                    const __bf16* __restrict__ Wbf,
                    __bf16* __restrict__ h1, __bf16* __restrict__ h2,
                    __bf16* __restrict__ Hf, float* __restrict__ Y,
                    float* __restrict__ out, int* cnt) {
  extern __shared__ __align__(16) char smem[];
  __bf16* lw = (__bf16*)smem;  // [5][32][512]

  const int tid = threadIdx.x;
  const int wg  = blockIdx.x;

  // ---- stage this WG's 32-row slice of each weight matrix into LDS ----
  // Wbf element offsets: [Whx0][Whx1][Whh0][Whh1][Wyh1]
  const int srcOff[5] = {0, 2 * WSZ, 1 * WSZ, 3 * WSZ, 4 * WSZ};
#pragma unroll
  for (int m = 0; m < 5; ++m) {
    const __bf16* src = Wbf + srcOff[m] + wg * 32 * Hh;
    __bf16* dst = lw + m * 32 * Hh;
    for (int v = tid; v < (32 * Hh) / 8; v += NTHREADS)
      *(v8bf*)(dst + v * 8) = *(const v8bf*)(src + v * 8);
  }
  __syncthreads();

  // ---- per-thread tile coordinates ----
  const int w     = tid >> 5;          // wave 0..7
  const int lane  = tid & 31;
  const int mtile = w & 3;             // 4 M-tiles cover B=64
  const int ntl   = w >> 2;            // 2 local N-tiles (32 cols / WG)
  const int mbase = mtile * 16;
  const int laneM = lane & 15;
  const int khalf = lane >> 4;
  const int mA    = mbase + laneM;     // A-operand row (batch index)
  const int nloc  = ntl * 16 + laneM;  // local column in WG slice
  const int nglob = wg * 32 + nloc;    // global hidden column

  const float bias1 = b_h[nglob];
  const float bias2 = b_h[Hh + nglob];
  const float biasy = b_y[Hh + nglob];

  int mr[8], sl[8];
#pragma unroll
  for (int r = 0; r < 8; ++r) {        // C/D layout: lanes<16 -> M=r, else M=8+r
    mr[r] = mbase + khalf * 8 + r;
    sl[r] = slen[mr[r]];
  }

  const __bf16* Bx1 = lw + 0 * 32 * Hh + nloc * Hh;
  const __bf16* Bh1 = lw + 1 * 32 * Hh + nloc * Hh;
  const __bf16* Bx2 = lw + 2 * 32 * Hh + nloc * Hh;
  const __bf16* Bh2 = lw + 3 * 32 * Hh + nloc * Hh;
  const __bf16* By  = lw + 4 * 32 * Hh + nloc * Hh;

  int gen = 0;
  for (int t = 0; t < Tt; ++t) {
    const int p = t & 1, q = p ^ 1;
    const __bf16* h1r = h1 + p * Bb * Hh;
    __bf16*       h1w = h1 + q * Bb * Hh;
    const __bf16* h2r = h2 + p * Bb * Hh;
    __bf16*       h2w = h2 + q * Bb * Hh;

    // ---------------- layer 1: h1 = tanh(x_t Whx0^T + h1 Whh0^T + b0) ------
    const float*  erow = emb + (size_t)x[(size_t)mA * Tt + t] * Ee;
    const __bf16* ah1  = h1r + mA * Hh;
    v8f c = {};
#pragma unroll
    for (int kb = 0; kb < 16; ++kb) {
      const int k0 = kb * 32;
      c = wmma_bf16(load_a_f32(erow, k0, khalf),  load_b_lds(Bx1, k0, khalf), c);
      c = wmma_bf16(load_a_bf16(ah1, k0, khalf),  load_b_lds(Bh1, k0, khalf), c);
    }
#pragma unroll
    for (int r = 0; r < 8; ++r)
      h1w[mr[r] * Hh + nglob] = (__bf16)tanhf(c[r] + bias1);

    ++gen;
    grid_barrier(cnt, gen * NWG);  // all columns of h1(t) visible

    // ---------------- layer 2: h2 = tanh(h1 Whx1^T + h2 Whh1^T + b1) -------
    const __bf16* ax2 = h1w + mA * Hh;
    const __bf16* ah2 = h2r + mA * Hh;
    c = (v8f){};
#pragma unroll
    for (int kb = 0; kb < 16; ++kb) {
      const int k0 = kb * 32;
      c = wmma_bf16(load_a_bf16(ax2, k0, khalf), load_b_lds(Bx2, k0, khalf), c);
      c = wmma_bf16(load_a_bf16(ah2, k0, khalf), load_b_lds(Bh2, k0, khalf), c);
    }
#pragma unroll
    for (int r = 0; r < 8; ++r) {
      float v = tanhf(c[r] + bias2);
      h2w[mr[r] * Hh + nglob] = (__bf16)v;
      if (t == sl[r] - 1) Hf[mr[r] * Hh + nglob] = (__bf16)v;  // snapshot
    }
  }

  ++gen;
  grid_barrier(cnt, gen * NWG);  // Hfinal complete

  // ---------------- Y = Hfinal @ Wyh[1]^T + b_y[1]  (per-WG column slice) --
  {
    const __bf16* aH = Hf + mA * Hh;
    v8f c = {};
#pragma unroll
    for (int kb = 0; kb < 16; ++kb) {
      const int k0 = kb * 32;
      c = wmma_bf16(load_a_bf16(aH, k0, khalf), load_b_lds(By, k0, khalf), c);
    }
#pragma unroll
    for (int r = 0; r < 8; ++r)
      Y[mr[r] * Hh + nglob] = c[r] + biasy;
  }

  ++gen;
  grid_barrier(cnt, gen * NWG);  // Y complete

  // ---------------- out = Y @ Wf^T + bf  (64x32, WG0, f32) -----------------
  if (wg == 0) {
    for (int idx = tid; idx < Bb * Oo; idx += NTHREADS) {
      const int b = idx >> 5, o = idx & 31;
      const float* yr = Y  + b * Hh;
      const float* wr = Wf + o * Hh;
      float acc = bfv[o];
      for (int k = 0; k < Hh; ++k) acc += yr[k] * wr[k];
      out[idx] = acc;
    }
  }
}

// ---------------------------------------------------------------------------
// Host launcher
// ---------------------------------------------------------------------------
extern "C" void kernel_launch(void* const* d_in, const int* in_sizes, int n_in,
                              void* d_out, int out_size, void* d_ws, size_t ws_size,
                              hipStream_t stream) {
  const int*   x    = (const int*)d_in[0];
  const int*   slen = (const int*)d_in[1];
  const float* emb  = (const float*)d_in[2];
  const float* Whx  = (const float*)d_in[3];
  const float* Whh  = (const float*)d_in[4];
  const float* b_h  = (const float*)d_in[5];
  const float* Wyh  = (const float*)d_in[6];
  const float* b_y  = (const float*)d_in[7];
  const float* Wf   = (const float*)d_in[8];
  const float* bfv  = (const float*)d_in[9];
  float* out = (float*)d_out;
  char*  ws  = (char*)d_ws;

  // workspace layout (all 256B-aligned)
  size_t o = 0;
  int*    cnt = (int*)(ws + o);    o += 256;
  __bf16* Wbf = (__bf16*)(ws + o); o += (size_t)5 * WSZ * 2;       // 2.62 MB
  __bf16* h1  = (__bf16*)(ws + o); o += (size_t)2 * Bb * Hh * 2;   // double buf
  __bf16* h2  = (__bf16*)(ws + o); o += (size_t)2 * Bb * Hh * 2;
  __bf16* Hf  = (__bf16*)(ws + o); o += (size_t)Bb * Hh * 2;
  float*  Y   = (float*)(ws + o);  o += (size_t)Bb * Hh * 4;

  // zero barrier counter + hidden-state double buffers (h0 = 0)
  zero_bytes16<<<1, 256, 0, stream>>>((char*)cnt, 256 / 16);
  zero_bytes16<<<64, 256, 0, stream>>>((char*)h1, (4 * Bb * Hh * 2) / 16);

  // weights -> bf16, row-major [out][in] (exactly the B-operand layout)
  cvt_f32_to_bf16<<<512, 256, 0, stream>>>(Whx,       Wbf + 0 * WSZ, 2 * WSZ);
  cvt_f32_to_bf16<<<512, 256, 0, stream>>>(Whh,       Wbf + 2 * WSZ, 2 * WSZ);
  cvt_f32_to_bf16<<<256, 256, 0, stream>>>(Wyh + WSZ, Wbf + 4 * WSZ, 1 * WSZ);

  const size_t lds_bytes = (size_t)5 * 32 * Hh * sizeof(__bf16);  // 160 KB
  rnn_persistent<<<dim3(NWG), dim3(NTHREADS), lds_bytes, stream>>>(
      x, slen, emb, b_h, b_y, Wf, bfv, Wbf, h1, h2, Hf, Y, out, cnt);
}
